// MyConv_27470610825753
// MI455X (gfx1250) — matrix operand order repaired and checked by reading, not compiled, not run on hardware
//
#include <hip/hip_runtime.h>

typedef __attribute__((ext_vector_type(16))) __bf16 v16bf;
typedef __attribute__((ext_vector_type(8)))  float  v8f;

#define CIN   128
#define COUT  128
#define HW    128
#define NB    8
#define KST   36      // k-steps of 32 (9 taps * 128ci / 32)
#define WTILE 64      // output pixels per workgroup

// ---------------------------------------------------------------------------
// Kernel 0: repack f32 weights [Cout][Cin][3][3] into bf16 WMMA A-fragments
// layout [mtile(8)][kstep(36)][lane(32)][elem(16)]  (288 KB, stays in L2).
// 16-bit A order used here (k contiguous per lane half): lanes 0-15 => M=lane, elems map to
// K = 0..15; lanes 16-31 => same M, K = 16..31.
// Global K enumeration: k = (kh*3+kw)*128 + ci ; kstep covers 32 consecutive k.
// ---------------------------------------------------------------------------
__global__ void wprep_kernel(const float* __restrict__ w,
                             __bf16* __restrict__ afrag) {
    int idx = blockIdx.x * blockDim.x + threadIdx.x;   // < 8*36*32*16 = 147456
    int e  = idx & 15;
    int l  = (idx >> 4) & 31;
    int ks = (idx >> 9) % KST;
    int mt = idx / (512 * KST);

    int kpos = ks >> 2;                 // 0..8  -> (kh,kw)
    int kh = kpos / 3, kw = kpos % 3;
    int kval = e + ((l >= 16) ? 16 : 0);   // 0..31
    int ci = ((ks & 3) << 5) + kval;
    int co = mt * 16 + (l & 15);

    afrag[idx] = (__bf16)w[((co * CIN + ci) * 3 + kh) * 3 + kw];
}

// ---------------------------------------------------------------------------
// Kernel 1: implicit-GEMM masked conv. One WG = 128 Cout x 64 pixels of 1 row.
// ---------------------------------------------------------------------------
union BF16V { v16bf v; uint4 u[2]; };

__global__ __launch_bounds__(256)
void mconv_kernel(const float* __restrict__ x,
                  const int*   __restrict__ mask,
                  const __bf16* __restrict__ afrag,
                  const float* __restrict__ bias,
                  float* __restrict__ out) {
    // x strip (3 rows x 66 cols halo) as bf16, ci padded 128->136 (16B steps)
    __shared__ __align__(32) __bf16 xs[3][66][136];
    __shared__ int ms[3][68];

    const int tid   = threadIdx.x;
    const int wg    = blockIdx.x;                // 8 * 128 * 2 = 2048
    const int wt    = wg & 1;
    const int h     = (wg >> 1) & (HW - 1);
    const int b     = wg >> 8;
    const int w0    = wt * WTILE;

    // ---- stage x (f32 -> bf16) into LDS, zero-padded halo ----
    const int TOT = 3 * 66 * CIN;
    for (int i = tid; i < TOT; i += 256) {
        int col  = i % 66;
        int rest = i / 66;
        int row  = rest % 3;
        int ci   = rest / 3;
        int ir = h - 1 + row;
        int ic = w0 - 1 + col;
        float v = 0.0f;
        if ((unsigned)ir < HW && (unsigned)ic < HW)
            v = x[(((b * CIN + ci) * HW) + ir) * HW + ic];
        xs[row][col][ci] = (__bf16)v;
    }
    // ---- stage mask strip ----
    if (tid < 3 * 66) {
        int col = tid % 66;
        int row = tid / 66;
        int ir = h - 1 + row;
        int ic = w0 - 1 + col;
        int mv = 0;
        if ((unsigned)ir < HW && (unsigned)ic < HW)
            mv = mask[(b * HW + ir) * HW + ic];
        ms[row][col] = mv;
    }
    __syncthreads();

    const int wave = tid >> 5;       // 8 waves: M-subtile (16 Cout each)
    const int lane = tid & 31;
    const int ln16 = lane & 15;
    const int hi   = lane >> 4;      // 0/1 : lane half

    v8f acc[4] = {};                 // 4 N-subtiles of 16 pixels

    const __bf16* abase = afrag + ((size_t)wave * KST * 32 + (size_t)lane) * 16;
    // Base pointers into the LDS strip for this lane (col = ln16 + kw later).
    const __bf16* xbase = &xs[0][ln16][hi << 4];

    // Fully unrolled k-loop: accumulators stay pinned (no loop-carried register
    // rotation -> no v_mov shuffles / hazard nops), all constant offsets fold
    // into ds_load/global_load immediates, and A loads issue far ahead.
#pragma unroll
    for (int ks = 0; ks < KST; ++ks) {
        const int kpos = ks >> 2;
        const int kh = kpos / 3, kw = kpos % 3;      // constants after unroll
        const int ci0 = (ks & 3) << 5;               // constant after unroll

        // A-fragment from L2-resident preswizzled weights
        v16bf a = *(const v16bf*)(abase + (size_t)ks * 512);
        if (ks + 1 < KST)
            __builtin_prefetch(abase + (size_t)(ks + 1) * 512, 0, 3);

#pragma unroll
        for (int nb = 0; nb < 4; ++nb) {
            // B fragment: lane -> pixel column (N = lane%16), 16 contiguous ci
            const __bf16* p = xbase + ((size_t)kh * 66 + nb * 16 + kw) * 136 + ci0;
            BF16V bb;
            bb.u[0] = *(const uint4*)p;
            bb.u[1] = *(const uint4*)(p + 8);
            acc[nb] = __builtin_amdgcn_wmma_f32_16x16x32_bf16(
                false, a, false, bb.v, (short)0, acc[nb], false, false);
        }
    }

    // ---- epilogue: bias + mask gating, full output coverage ----
#pragma unroll
    for (int nb = 0; nb < 4; ++nb) {
        const int pn = nb * 16 + ln16;       // pixel within tile
        const int wcol = w0 + pn;
        int act = 0;
#pragma unroll
        for (int r = 0; r < 3; ++r)
#pragma unroll
            for (int c = 0; c < 3; ++c)
                act |= ms[r][pn + c];
        const float gate = act ? 1.0f : 0.0f;

#pragma unroll
        for (int r = 0; r < 8; ++r) {
            // C/D layout: VGPR r, lanes 0-15 -> M=r ; lanes 16-31 -> M=r+8
            const int co = wave * 16 + r + hi * 8;
            const float val = (acc[nb][r] + bias[co]) * gate;
            out[(((b * COUT + co) * HW) + h) * HW + wcol] = val;
        }
    }
}

// ---------------------------------------------------------------------------
extern "C" void kernel_launch(void* const* d_in, const int* in_sizes, int n_in,
                              void* d_out, int out_size, void* d_ws, size_t ws_size,
                              hipStream_t stream) {
    const float* x    = (const float*)d_in[0];
    const int*   mask = (const int*)  d_in[1];
    const float* w    = (const float*)d_in[2];
    const float* bias = (const float*)d_in[3];
    float* out = (float*)d_out;

    __bf16* afrag = (__bf16*)d_ws;   // 8*36*32*16 bf16 = 294,912 bytes

    const int prep_elems = 8 * KST * 32 * 16;           // 147456
    wprep_kernel<<<prep_elems / 256, 256, 0, stream>>>(w, afrag);

    const int nblocks = NB * HW * (HW / WTILE);          // 2048
    mconv_kernel<<<nblocks, 256, 0, stream>>>(x, mask, afrag, bias, out);
}
